// ImplicitNetworkGrid_85676007620807
// MI455X (gfx1250) — matrix operand order repaired and checked by reading, not compiled, not run on hardware
//
#include <hip/hip_runtime.h>
#include <hip/hip_bf16.h>

#define NPTS 524288
#define TILE 128
#define LDH  256

typedef __attribute__((ext_vector_type(16))) _Float16 v16h;
typedef __attribute__((ext_vector_type(8)))  _Float16 v8h;
typedef __attribute__((ext_vector_type(8)))  float    v8f;

// Hash-grid level metadata (computed from reference: base 16 -> 2048, 16 levels,
// scale = 2^(7/15), n_level = ceil(min(2^19,(R+1)^3)/8)*8, levels >=5 hashed).
__constant__ int      c_res[16] = {16,23,31,43,59,81,112,154,213,295,407,562,777,1073,1483,2048};
__constant__ unsigned c_off[16] = {0u,4920u,18744u,51512u,136696u,352696u,876984u,1401272u,
                                   1925560u,2449848u,2974136u,3498424u,4022712u,4547000u,
                                   5071288u,5595576u};

// Branchless softplus100: ln(1+e^{100x})/100 = max(x,0) + ln(1+e^{-|100x|})/100.
// Raw v_exp_f32 / v_log_f32 (base-2); log argument is in (1,2] -> no fixups.
static __device__ __forceinline__ float softplus100(float x) {
  const float NEG100LOG2E = -144.26950408889634f;   // -100 * log2(e)
  const float LN2_OVER100 = 0.006931471805599453f;  // ln(2) / 100
  float e = __builtin_amdgcn_exp2f(fabsf(x) * NEG100LOG2E);  // e^{-|100x|} in (0,1]
  float l = __builtin_amdgcn_logf(1.f + e);                  // log2(1+e) in (0,1]
  return fmaxf(x, 0.f) + l * LN2_OVER100;
}

static __device__ __forceinline__ v16h cat8(v8h lo, v8h hi) {
  return __builtin_shufflevector(lo, hi, 0,1,2,3,4,5,6,7,8,9,10,11,12,13,14,15);
}

// ---------------------------------------------------------------------------
// Weight-norm prep: Wt[k][o] = g[o] * v[o][k] / ||v[o]||, f16, transposed+padded.
// grid = 769 blocks (256 rows L0, 256 rows L1, 257 rows L2), block = 256.
// ---------------------------------------------------------------------------
__global__ void __launch_bounds__(256)
wn_prep_kernel(const float* __restrict__ v0, const float* __restrict__ g0,
               const float* __restrict__ v1, const float* __restrict__ g1,
               const float* __restrict__ v2, const float* __restrict__ g2,
               _Float16* __restrict__ W0t, _Float16* __restrict__ W1t,
               _Float16* __restrict__ W2t) {
  __shared__ float red[256];
  const int b = blockIdx.x, t = threadIdx.x;
  const float* vrow;
  int nin, o;
  if (b < 256)      { o = b;       vrow = v0 + o * 35;  nin = 35;  }
  else if (b < 512) { o = b - 256; vrow = v1 + o * 256; nin = 256; }
  else              { o = b - 512; vrow = v2 + o * 256; nin = 256; }

  float s = 0.f;
  for (int i = t; i < nin; i += 256) { float vv = vrow[i]; s += vv * vv; }
  red[t] = s;
  __syncthreads();
  for (int k = 128; k > 0; k >>= 1) { if (t < k) red[t] += red[t + k]; __syncthreads(); }

  float scale;
  if (b < 256)      scale = g0[o] * rsqrtf(red[0]);
  else if (b < 512) scale = g1[o] * rsqrtf(red[0]);
  else              scale = g2[o] * rsqrtf(red[0]);

  if (b < 256) {
    if (t < 64) W0t[t * 256 + o] = (t < 35) ? (_Float16)(vrow[t] * scale) : (_Float16)0.f;
  } else if (b < 512) {
    W1t[t * 256 + o] = (_Float16)(vrow[t] * scale);
  } else {
    W2t[t * 272 + o] = (_Float16)(vrow[t] * scale);
    if (o == 0) {  // zero O-padding cols 257..271 for all K rows
      for (int i = t; i < 256 * 15; i += 256) {
        int k = i / 15, c = 257 + (i % 15);
        W2t[k * 272 + c] = (_Float16)0.f;
      }
    }
  }
}

// ---------------------------------------------------------------------------
// Fused: hash-grid encode -> 3-layer weight-norm MLP via WMMA, per 128-row tile.
// 256 threads = 8 waves; each wave owns 16 rows; activations live in 64KB LDS.
// ---------------------------------------------------------------------------
__global__ void __launch_bounds__(256)
implicit_net_fused_kernel(const float* __restrict__ x, const float* __restrict__ table,
                          const _Float16* __restrict__ W0t, const float* __restrict__ b0,
                          const _Float16* __restrict__ W1t, const float* __restrict__ b1,
                          const _Float16* __restrict__ W2t, const float* __restrict__ b2,
                          float* __restrict__ out) {
  __shared__ _Float16 sH[TILE * LDH];  // exactly 64 KB

  const int tid  = threadIdx.x;
  const int base = blockIdx.x * TILE;

  // ---------------- phase 1: hash-grid encoding (2 threads per point) -------
  {
    const int p = tid & (TILE - 1);
    const int half = tid >> 7;       // 0: levels 0-7 + x, 1: levels 8-15 + pad
    const int n = base + p;
    // x is read exactly once per point across the dispatch -> non-temporal.
    float xv[3];
    xv[0] = __builtin_nontemporal_load(x + 3 * n + 0);
    xv[1] = __builtin_nontemporal_load(x + 3 * n + 1);
    xv[2] = __builtin_nontemporal_load(x + 3 * n + 2);
    _Float16* hrow = sH + p * LDH;
    if (half == 0) {
      hrow[0] = (_Float16)xv[0]; hrow[1] = (_Float16)xv[1]; hrow[2] = (_Float16)xv[2];
    } else {
      #pragma unroll
      for (int c = 35; c < 64; ++c) hrow[c] = (_Float16)0.f;  // K padding
    }
    float x01[3];
    #pragma unroll
    for (int d = 0; d < 3; ++d) {
      float v = (xv[d] * (1.f / 1.5f) + 1.f) * 0.5f;
      x01[d] = fminf(fmaxf(v, 0.f), 1.f);
    }
    const int l0 = half * 8;
    for (int l = l0; l < l0 + 8; ++l) {
      const int R = c_res[l];
      const unsigned off = c_off[l];
      const unsigned S = (unsigned)(R + 1);
      unsigned i0[3]; float fr[3];
      #pragma unroll
      for (int d = 0; d < 3; ++d) {
        float pos = x01[d] * (float)(R - 1);
        float fl = floorf(pos);
        fl = fminf(fmaxf(fl, 0.f), (float)(R - 2));
        i0[d] = (unsigned)fl;
        fr[d] = pos - fl;
      }
      float f0 = 0.f, f1 = 0.f;
      #pragma unroll
      for (int c = 0; c < 8; ++c) {
        unsigned cx = i0[0] + (c & 1);
        unsigned cy = i0[1] + ((c >> 1) & 1);
        unsigned cz = i0[2] + ((c >> 2) & 1);
        float w = ((c & 1) ? fr[0] : 1.f - fr[0])
                * ((c & 2) ? fr[1] : 1.f - fr[1])
                * ((c & 4) ? fr[2] : 1.f - fr[2]);
        unsigned idx;
        if (l >= 5) {  // hashed levels; n_level == 2^19 so '& mask' == mod
          idx = (cx ^ (cy * 2654435761u) ^ (cz * 805459861u)) & (524288u - 1u);
        } else {       // dense levels; max idx < n_level by construction
          idx = cx + cy * S + cz * S * S;
        }
        const float* te = table + (size_t)(idx + off) * 2;  // L2-resident reuse set
        f0 += w * te[0];
        f1 += w * te[1];
      }
      hrow[3 + 2 * l]     = (_Float16)f0;
      hrow[3 + 2 * l + 1] = (_Float16)f1;
    }
  }
  __syncthreads();

  // ---------------- phase 2: WMMA MLP ---------------------------------------
  const int lane = tid & 31;
  const int wave = tid >> 5;
  const int rowBase = wave * 16;      // each wave owns 16 consecutive rows
  const int m  = lane & 15;           // A row / D col within tile
  const int hi = lane >> 4;           // lane half-group
  const _Float16* arow = sH + (rowBase + m) * LDH;

  // ---- layer 0: K=64 (2 ktiles), O=256 (16 otiles), softplus100, in-place --
  {
    v16h A[2];
    #pragma unroll
    for (int kt = 0; kt < 2; ++kt) {
      v8h lo = *(const v8h*)(arow + kt * 32 + hi * 8);        // K = kt*32+hi*8+0..7
      v8h hh = *(const v8h*)(arow + kt * 32 + 16 + hi * 8);   // K = +16
      A[kt] = cat8(lo, hh);
    }
    __syncthreads();  // all A (cols 0..63) read before in-place overwrite
    for (int ot = 0; ot < 16; ++ot) {
      v8f acc = {};
      #pragma unroll
      for (int kt = 0; kt < 2; ++kt) {
        const _Float16* bp = W0t + (size_t)(kt * 32 + hi * 8) * 256 + ot * 16 + m;
        v16h B;
        #pragma unroll
        for (int s = 0; s < 8; ++s) { B[s] = bp[s * 256]; B[8 + s] = bp[(16 + s) * 256]; }
        acc = __builtin_amdgcn_wmma_f32_16x16x32_f16(false, A[kt], false, B,
                                                     (short)0, acc, false, false);
      }
      const float bias = b0[ot * 16 + m];
      #pragma unroll
      for (int r = 0; r < 8; ++r) {
        float vl = softplus100(acc[r] + bias);
        sH[(rowBase + r + 8 * hi) * LDH + ot * 16 + m] = (_Float16)vl;
      }
    }
  }
  __syncthreads();

  // ---- layer 1: K=256 (8 ktiles, A hoisted to regs), O=256, softplus, in-place
  {
    v16h A[8];
    #pragma unroll
    for (int kt = 0; kt < 8; ++kt) {
      v8h lo = *(const v8h*)(arow + kt * 32 + hi * 8);
      v8h hh = *(const v8h*)(arow + kt * 32 + 16 + hi * 8);
      A[kt] = cat8(lo, hh);
    }
    __syncthreads();
    for (int ot = 0; ot < 16; ++ot) {
      v8f acc = {};
      #pragma unroll
      for (int kt = 0; kt < 8; ++kt) {
        const _Float16* bp = W1t + (size_t)(kt * 32 + hi * 8) * 256 + ot * 16 + m;
        v16h B;
        #pragma unroll
        for (int s = 0; s < 8; ++s) { B[s] = bp[s * 256]; B[8 + s] = bp[(16 + s) * 256]; }
        acc = __builtin_amdgcn_wmma_f32_16x16x32_f16(false, A[kt], false, B,
                                                     (short)0, acc, false, false);
      }
      const float bias = b1[ot * 16 + m];
      #pragma unroll
      for (int r = 0; r < 8; ++r) {
        float vl = softplus100(acc[r] + bias);
        sH[(rowBase + r + 8 * hi) * LDH + ot * 16 + m] = (_Float16)vl;
      }
    }
  }
  __syncthreads();

  // ---- layer 2: K=256, O=257 (17 otiles, padded to 272), f32 NT store ------
  {
    v16h A[8];
    #pragma unroll
    for (int kt = 0; kt < 8; ++kt) {
      v8h lo = *(const v8h*)(arow + kt * 32 + hi * 8);
      v8h hh = *(const v8h*)(arow + kt * 32 + 16 + hi * 8);
      A[kt] = cat8(lo, hh);
    }
    for (int ot = 0; ot < 17; ++ot) {
      v8f acc = {};
      #pragma unroll
      for (int kt = 0; kt < 8; ++kt) {
        const _Float16* bp = W2t + (size_t)(kt * 32 + hi * 8) * 272 + ot * 16 + m;
        v16h B;
        #pragma unroll
        for (int s = 0; s < 8; ++s) { B[s] = bp[s * 272]; B[8 + s] = bp[(16 + s) * 272]; }
        acc = __builtin_amdgcn_wmma_f32_16x16x32_f16(false, A[kt], false, B,
                                                     (short)0, acc, false, false);
      }
      const int col = ot * 16 + m;   // guard AFTER wmma so EXEC==all-ones for wmma
      if (col < 257) {
        const float bias = b2[col];
        // 539 MB write-once stream: non-temporal so it doesn't evict the
        // L2-resident hash table / weights.
        #pragma unroll
        for (int r = 0; r < 8; ++r) {
          const int n = base + rowBase + r + 8 * hi;
          __builtin_nontemporal_store(acc[r] + bias, out + (size_t)n * 257 + col);
        }
      }
    }
  }
}

// ---------------------------------------------------------------------------
extern "C" void kernel_launch(void* const* d_in, const int* in_sizes, int n_in,
                              void* d_out, int out_size, void* d_ws, size_t ws_size,
                              hipStream_t stream) {
  const float* x   = (const float*)d_in[0];
  const float* tab = (const float*)d_in[1];
  const float* v0  = (const float*)d_in[2];
  const float* g0  = (const float*)d_in[3];
  const float* b0  = (const float*)d_in[4];
  const float* v1  = (const float*)d_in[5];
  const float* g1  = (const float*)d_in[6];
  const float* b1  = (const float*)d_in[7];
  const float* v2  = (const float*)d_in[8];
  const float* g2  = (const float*)d_in[9];
  const float* b2  = (const float*)d_in[10];
  float* out = (float*)d_out;

  char* ws = (char*)d_ws;
  _Float16* W0t = (_Float16*)(ws);                       // 64*256*2  = 32768 B
  _Float16* W1t = (_Float16*)(ws + 32768);               // 256*256*2 = 131072 B
  _Float16* W2t = (_Float16*)(ws + 32768 + 131072);      // 256*272*2 = 139264 B

  wn_prep_kernel<<<769, 256, 0, stream>>>(v0, g0, v1, g1, v2, g2, W0t, W1t, W2t);
  implicit_net_fused_kernel<<<NPTS / TILE, 256, 0, stream>>>(
      x, tab, W0t, b0, W1t, b1, W2t, b2, out);
}
